// Transformer1_20761871909261
// MI455X (gfx1250) — compile-verified, matrix operand order checked by reading, not run on hardware
//
#include <hip/hip_runtime.h>
#include <hip/hip_bf16.h>

// ---------------------------------------------------------------------------
// Types for WMMA fragments
// ---------------------------------------------------------------------------
typedef __attribute__((ext_vector_type(16))) __bf16 v16bf;
typedef __attribute__((ext_vector_type(8)))  float  v8f;

struct alignas(16) U4 { unsigned int x, y, z, w; };
union Frag { U4 u[2]; v16bf v; };

// TDM descriptor vector types (amdgpu-toolchain clang-23 6-arg form)
typedef unsigned int tdm_v4u __attribute__((ext_vector_type(4)));
typedef int          tdm_v8i __attribute__((ext_vector_type(8)));
typedef int          tdm_v4i __attribute__((ext_vector_type(4)));

#if defined(__gfx1250__) && __has_builtin(__builtin_amdgcn_tensor_load_to_lds)
#define USE_TDM 1
#else
#define USE_TDM 0
#endif

// Model dims
#define BB   2
#define TT   2048
#define DD   1024
#define HH   4096
#define VV   50257
#define NTOK (BB * TT)

// ---------------------------------------------------------------------------
// f32 -> bf16 conversion (grid-stride)
// ---------------------------------------------------------------------------
__global__ void tf1_cvt_bf16(const float* __restrict__ in, __bf16* __restrict__ out,
                             size_t n) {
    size_t i = (size_t)blockIdx.x * blockDim.x + threadIdx.x;
    size_t stride = (size_t)gridDim.x * blockDim.x;
    for (; i < n; i += stride) out[i] = (__bf16)in[i];
}

// ---------------------------------------------------------------------------
// Embedding gather -> bf16 activations
// ---------------------------------------------------------------------------
__global__ void tf1_embed_bf16(const int* __restrict__ tokens,
                               const float* __restrict__ embW,
                               __bf16* __restrict__ x, int ntok, int d) {
    size_t total = (size_t)ntok * d;
    size_t i = (size_t)blockIdx.x * blockDim.x + threadIdx.x;
    size_t stride = (size_t)gridDim.x * blockDim.x;
    for (; i < total; i += stride) {
        int n = (int)(i / d);
        int c = (int)(i % d);
        x[i] = (__bf16)embW[(size_t)tokens[n] * d + c];
    }
}

// ---------------------------------------------------------------------------
// bf16 transpose (per batch): out[b][d][k] = in[b][k][d]
// ---------------------------------------------------------------------------
__global__ void tf1_transpose_bf16(const __bf16* __restrict__ in,
                                   __bf16* __restrict__ out,
                                   int nb, int rows, int cols) {
    size_t total = (size_t)nb * rows * cols;
    size_t i = (size_t)blockIdx.x * blockDim.x + threadIdx.x;
    size_t stride = (size_t)gridDim.x * blockDim.x;
    for (; i < total; i += stride) {
        size_t b   = i / ((size_t)rows * cols);
        size_t rem = i % ((size_t)rows * cols);
        int k = (int)(rem / cols);
        int d = (int)(rem % cols);
        out[(b * cols + d) * rows + k] = in[i];
    }
}

// ---------------------------------------------------------------------------
// Causal softmax over f32 scores -> bf16 attention weights
// ---------------------------------------------------------------------------
__global__ void tf1_softmax_causal(const float* __restrict__ scores,
                                   __bf16* __restrict__ attn,
                                   int tlen, float scale) {
    int row = blockIdx.x;                 // 0 .. B*T-1
    int q = row % tlen;
    const float* s = scores + (size_t)row * tlen;
    __bf16* a = attn + (size_t)row * tlen;
    int nvalid = q + 1;

    __shared__ float red[256];
    int tid = threadIdx.x;

    float m = -3.4e38f;
    for (int k = tid; k < nvalid; k += 256) m = fmaxf(m, s[k] * scale);
    red[tid] = m; __syncthreads();
    for (int st = 128; st > 0; st >>= 1) {
        if (tid < st) red[tid] = fmaxf(red[tid], red[tid + st]);
        __syncthreads();
    }
    m = red[0]; __syncthreads();

    float sum = 0.0f;
    for (int k = tid; k < nvalid; k += 256) sum += __expf(s[k] * scale - m);
    red[tid] = sum; __syncthreads();
    for (int st = 128; st > 0; st >>= 1) {
        if (tid < st) red[tid] += red[tid + st];
        __syncthreads();
    }
    sum = red[0];
    float inv = 1.0f / sum;

    for (int k = tid; k < tlen; k += 256) {
        float v = (k < nvalid) ? __expf(s[k] * scale - m) * inv : 0.0f;
        a[k] = (__bf16)v;
    }
}

// ---------------------------------------------------------------------------
// Tiled WMMA GEMM:  C[M,N] = A[M,K] * B[N,K]^T (+ bias[N]) (+ ReLU)
// Block = 128 threads = 4 wave32; each wave computes a 64x64 sub-tile as
// 4x4 v_wmma_f32_16x16x32_bf16 accumulators. LDS tiles (80B row stride,
// bank-conflict free) are double-buffered and filled by the Tensor Data
// Mover (TENSOR_LOAD_TO_LDS, wave 0 issues, overlap with compute, ordered
// by s_wait_tensorcnt) with a plain global->LDS staging fallback.
// M multiple of 128; K multiple of 32; ragged N handled by tensor_dim /
// store guards when NGUARD.
// ---------------------------------------------------------------------------
#define BM 128
#define BN 128
#define BK 32
#define LDSROW 40   // 32 bf16 data + 8 bf16 pad -> 80B row stride

#if USE_TDM
// Issue one TDM 2D tile load: 128 rows x 32 bf16 from row-major src
// (ld = ldK elems) into LDS at ldsOff, padding +16B after every 64B row.
__device__ __forceinline__ void tf1_tdm_issue(const __bf16* src, unsigned ldsOff,
                                              unsigned rowsRemain, int kRemain,
                                              unsigned ldK) {
    unsigned long long ga = (unsigned long long)(size_t)src;
    tdm_v4u g0;
    g0[0] = 1u;                                         // count=1, user descriptor
    g0[1] = ldsOff;                                     // lds_addr (bytes)
    g0[2] = (unsigned)(ga & 0xffffffffu);               // global_addr[31:0]
    g0[3] = (unsigned)((ga >> 32) & 0x1ffffffu) | (2u << 30);  // [56:32] | type=2
    unsigned td0 = (unsigned)kRemain;                   // tensor_dim0 (elems)
    unsigned td1 = rowsRemain;                          // tensor_dim1 (rows)
    tdm_v8i g1;
    g1[0] = (int)((1u << 16)      // data_size = 2 bytes
                | (1u << 20)      // pad_enable
                | (3u << 22)      // pad_interval: every 16 dwords (64B)
                | (3u << 25));    // pad_amount: 4 dwords (16B)
    g1[1] = (int)((td0 & 0xffffu) << 16);                       // tensor_dim0 lo
    g1[2] = (int)(((td0 >> 16) & 0xffffu) | ((td1 & 0xffffu) << 16));
    g1[3] = (int)(((td1 >> 16) & 0xffffu) | (32u << 16));       // tile_dim0 = 32
    g1[4] = (int)128;                                           // tile_dim1 = 128
    g1[5] = (int)ldK;                                           // dim0 stride lo
    g1[6] = 0;
    g1[7] = 0;
    tdm_v4i z4 = {0, 0, 0, 0};
    tdm_v8i z8 = {0, 0, 0, 0, 0, 0, 0, 0};
    __builtin_amdgcn_tensor_load_to_lds(g0, g1, z4, z4, z8, 0);
}
#endif

template <bool OUT_BF16, bool RELU, bool NGUARD>
__global__ __launch_bounds__(128)
void tf1_gemm_bf16(const __bf16* __restrict__ A, const __bf16* __restrict__ Bw,
                   const float* __restrict__ bias, void* __restrict__ Cout,
                   int M, int N, int K,
                   long long aStride, long long bStride, long long cStride) {
    __shared__ __attribute__((aligned(128))) __bf16 sA[2][BM * LDSROW];
    __shared__ __attribute__((aligned(128))) __bf16 sB[2][BN * LDSROW];

    const int tid  = threadIdx.x;
    const int bm   = blockIdx.y * BM;
    const int bn   = blockIdx.x * BN;
    const int bat  = blockIdx.z;

    A  += (size_t)bat * aStride;
    Bw += (size_t)bat * bStride;

    const int wave = tid >> 5;
    const int lane = tid & 31;
    const int wm   = (wave & 1) * 64;   // wave row offset in block tile
    const int wn   = (wave >> 1) * 64;  // wave col offset in block tile
    const int half = lane >> 4;         // K-half selector
    const int r    = lane & 15;         // row within 16

    v8f acc[4][4];
    const v8f vzero = {0.f, 0.f, 0.f, 0.f, 0.f, 0.f, 0.f, 0.f};
    #pragma unroll
    for (int i = 0; i < 4; ++i)
        #pragma unroll
        for (int j = 0; j < 4; ++j) acc[i][j] = vzero;

    const int nk = K / BK;

    // ---- staging (TDM or fallback) ----
#if USE_TDM
    const unsigned ldsA0 = (unsigned)(size_t)&sA[0][0];
    const unsigned ldsA1 = (unsigned)(size_t)&sA[1][0];
    const unsigned ldsB0 = (unsigned)(size_t)&sB[0][0];
    const unsigned ldsB1 = (unsigned)(size_t)&sB[1][0];
    auto stage = [&](int chunk, int buf) {
        if (tid < 32) {   // wave 0 issues the DMA descriptors
            int k0 = chunk * BK;
            tf1_tdm_issue(A + (size_t)bm * K + k0, buf ? ldsA1 : ldsA0,
                          (unsigned)(M - bm), K - k0, (unsigned)K);
            tf1_tdm_issue(Bw + (size_t)bn * K + k0, buf ? ldsB1 : ldsB0,
                          (unsigned)(N - bn), K - k0, (unsigned)K);
        }
    };
#else
    auto stage = [&](int chunk, int buf) {
        int k0 = chunk * BK;
        #pragma unroll
        for (int it = 0; it < 4; ++it) {
            int s   = tid + it * 128;
            int row = s >> 2;
            int seg = s & 3;
            const U4* src = (const U4*)(A + (size_t)(bm + row) * K + k0 + seg * 8);
            *(U4*)(&sA[buf][row * LDSROW + seg * 8]) = *src;
        }
        #pragma unroll
        for (int it = 0; it < 4; ++it) {
            int s   = tid + it * 128;
            int row = s >> 2;
            int seg = s & 3;
            U4 val; val.x = 0u; val.y = 0u; val.z = 0u; val.w = 0u;
            if (!NGUARD || (bn + row) < N)
                val = *(const U4*)(Bw + (size_t)(bn + row) * K + k0 + seg * 8);
            *(U4*)(&sB[buf][row * LDSROW + seg * 8]) = val;
        }
    };
#endif

    stage(0, 0);   // prologue: chunk 0 -> buffer 0

    for (int ic = 0; ic < nk; ++ic) {
        const int buf = ic & 1;
#if USE_TDM
        // chunk `ic` was issued after the previous barrier (chunk 0 in the prologue); wait for it.
#if __has_builtin(__builtin_amdgcn_s_wait_tensorcnt)
        __builtin_amdgcn_s_wait_tensorcnt(0);
#else
        asm volatile("s_wait_tensorcnt 0x0" ::: "memory");
#endif
#endif
        __syncthreads();                       // chunk ic visible; compute ic-1 done
        if (ic + 1 < nk) stage(ic + 1, buf ^ 1);   // overlap next DMA with compute

        const __bf16* la = sA[buf];
        const __bf16* lb = sB[buf];

        Frag fb[4];
        #pragma unroll
        for (int j = 0; j < 4; ++j) {
            int off = (wn + j * 16 + r) * LDSROW + half * 16;
            fb[j].u[0] = *(const U4*)(&lb[off]);
            fb[j].u[1] = *(const U4*)(&lb[off + 8]);
        }
        #pragma unroll
        for (int i = 0; i < 4; ++i) {
            Frag fa;
            int off = (wm + i * 16 + r) * LDSROW + half * 16;
            fa.u[0] = *(const U4*)(&la[off]);
            fa.u[1] = *(const U4*)(&la[off + 8]);
            #pragma unroll
            for (int j = 0; j < 4; ++j) {
                acc[i][j] = __builtin_amdgcn_wmma_f32_16x16x32_bf16(
                    false, fa.v, false, fb[j].v, (short)0, acc[i][j], false, false);
            }
        }
    }

    // ---- writeback: C/D layout: VGPR g -> M = g + 8*half, N = lane&15 ----
    #pragma unroll
    for (int i = 0; i < 4; ++i) {
        #pragma unroll
        for (int j = 0; j < 4; ++j) {
            int gm0 = bm + wm + i * 16 + half * 8;
            int gn  = bn + wn + j * 16 + r;
            if (NGUARD && gn >= N) continue;
            float badd = (bias != nullptr) ? bias[gn] : 0.0f;
            #pragma unroll
            for (int g = 0; g < 8; ++g) {
                float v = acc[i][j][g] + badd;
                if (RELU) v = fmaxf(v, 0.0f);
                size_t idx = (size_t)bat * cStride + (size_t)(gm0 + g) * N + gn;
                if (OUT_BF16) ((__bf16*)Cout)[idx] = (__bf16)v;
                else          ((float*)Cout)[idx]  = v;
            }
        }
    }
}

// ---------------------------------------------------------------------------
// Host-side launch helpers
// ---------------------------------------------------------------------------
template <bool OB, bool RL, bool NG>
static void run_gemm(const __bf16* A, const __bf16* Bw, const float* bias, void* C,
                     int M, int N, int K, int batch,
                     long long as, long long bs, long long cs, hipStream_t st) {
    dim3 grid((N + BN - 1) / BN, M / BM, batch);
    tf1_gemm_bf16<OB, RL, NG><<<grid, 128, 0, st>>>(A, Bw, bias, C, M, N, K, as, bs, cs);
}

extern "C" void kernel_launch(void* const* d_in, const int* in_sizes, int n_in,
                              void* d_out, int out_size, void* d_ws, size_t ws_size,
                              hipStream_t stream) {
    (void)in_sizes; (void)n_in; (void)out_size; (void)ws_size;

    const int*   tokens = (const int*)  d_in[0];
    const float* embW   = (const float*)d_in[1];
    const float* Wq     = (const float*)d_in[2];
    const float* Wk     = (const float*)d_in[3];
    const float* Wv     = (const float*)d_in[4];
    const float* Wo     = (const float*)d_in[5];
    const float* W1     = (const float*)d_in[6];
    const float* b1     = (const float*)d_in[7];
    const float* W2     = (const float*)d_in[8];
    const float* b2     = (const float*)d_in[9];
    const float* Wu     = (const float*)d_in[10];
    const float* bu     = (const float*)d_in[11];
    float* logits       = (float*)d_out;

    // ---- workspace carve-up (256B aligned) ----
    char* ws = (char*)d_ws;
    size_t off = 0;
    auto carve = [&](size_t bytes) -> char* {
        char* p = ws + off;
        off = (off + bytes + 255) & ~(size_t)255;
        return p;
    };
    __bf16* wq_b  = (__bf16*)carve((size_t)DD * DD * 2);
    __bf16* wk_b  = (__bf16*)carve((size_t)DD * DD * 2);
    __bf16* wv_b  = (__bf16*)carve((size_t)DD * DD * 2);
    __bf16* wo_b  = (__bf16*)carve((size_t)DD * DD * 2);
    __bf16* w1_b  = (__bf16*)carve((size_t)HH * DD * 2);
    __bf16* w2_b  = (__bf16*)carve((size_t)DD * HH * 2);
    __bf16* wu_b  = (__bf16*)carve((size_t)VV * DD * 2);
    __bf16* x_b   = (__bf16*)carve((size_t)NTOK * DD * 2);
    __bf16* q_b   = (__bf16*)carve((size_t)NTOK * DD * 2);
    __bf16* k_b   = (__bf16*)carve((size_t)NTOK * DD * 2);
    __bf16* v_b   = (__bf16*)carve((size_t)NTOK * DD * 2);
    float*  sc_f  = (float*) carve((size_t)BB * TT * TT * 4);
    __bf16* at_b  = (__bf16*)carve((size_t)BB * TT * TT * 2);
    __bf16* vt_b  = (__bf16*)carve((size_t)BB * DD * TT * 2);
    __bf16* ao_b  = (__bf16*)carve((size_t)NTOK * DD * 2);
    __bf16* xo_b  = (__bf16*)carve((size_t)NTOK * DD * 2);
    __bf16* h_b   = (__bf16*)carve((size_t)NTOK * HH * 2);
    __bf16* x2_b  = (__bf16*)carve((size_t)NTOK * DD * 2);

    // ---- weight conversion f32 -> bf16 ----
    tf1_cvt_bf16<<<2048, 256, 0, stream>>>(Wq, wq_b, (size_t)DD * DD);
    tf1_cvt_bf16<<<2048, 256, 0, stream>>>(Wk, wk_b, (size_t)DD * DD);
    tf1_cvt_bf16<<<2048, 256, 0, stream>>>(Wv, wv_b, (size_t)DD * DD);
    tf1_cvt_bf16<<<2048, 256, 0, stream>>>(Wo, wo_b, (size_t)DD * DD);
    tf1_cvt_bf16<<<4096, 256, 0, stream>>>(W1, w1_b, (size_t)HH * DD);
    tf1_cvt_bf16<<<4096, 256, 0, stream>>>(W2, w2_b, (size_t)DD * HH);
    tf1_cvt_bf16<<<8192, 256, 0, stream>>>(Wu, wu_b, (size_t)VV * DD);

    // ---- embedding gather ----
    tf1_embed_bf16<<<4096, 256, 0, stream>>>(tokens, embW, x_b, NTOK, DD);

    // ---- Q, K, V projections: [4096,1024] = x @ W^T ----
    run_gemm<true, false, false>(x_b, wq_b, nullptr, q_b, NTOK, DD, DD, 1, 0, 0, 0, stream);
    run_gemm<true, false, false>(x_b, wk_b, nullptr, k_b, NTOK, DD, DD, 1, 0, 0, 0, stream);
    run_gemm<true, false, false>(x_b, wv_b, nullptr, v_b, NTOK, DD, DD, 1, 0, 0, 0, stream);

    // ---- attention scores: per batch, [2048,2048] = Q @ K^T (f32 out) ----
    run_gemm<false, false, false>(q_b, k_b, nullptr, sc_f, TT, TT, DD, BB,
                                  (long long)TT * DD, (long long)TT * DD,
                                  (long long)TT * TT, stream);

    // ---- causal softmax (scale = 1/sqrt(1024)) -> bf16 attn ----
    tf1_softmax_causal<<<BB * TT, 256, 0, stream>>>(sc_f, at_b, TT, 0.03125f);

    // ---- transpose V per batch: Vt[b][d][k] ----
    tf1_transpose_bf16<<<4096, 256, 0, stream>>>(v_b, vt_b, BB, TT, DD);

    // ---- attn @ V: per batch, [2048,1024] = attn @ Vt^T ----
    run_gemm<true, false, false>(at_b, vt_b, nullptr, ao_b, TT, DD, TT, BB,
                                 (long long)TT * TT, (long long)DD * TT,
                                 (long long)TT * DD, stream);

    // ---- output projection: x = out @ Wo^T ----
    run_gemm<true, false, false>(ao_b, wo_b, nullptr, xo_b, NTOK, DD, DD, 1, 0, 0, 0, stream);

    // ---- MLP: h = relu(x @ W1^T + b1); x2 = h @ W2^T + b2 ----
    run_gemm<true, true,  false>(xo_b, w1_b, b1, h_b,  NTOK, HH, DD, 1, 0, 0, 0, stream);
    run_gemm<true, false, false>(h_b,  w2_b, b2, x2_b, NTOK, DD, HH, 1, 0, 0, 0, stream);

    // ---- unembed: logits[4096,50257] = x2 @ Wu^T + bu (f32, N-guarded) ----
    run_gemm<false, false, true>(x2_b, wu_b, bu, logits, NTOK, VV, DD, 1, 0, 0, 0, stream);
}
